// LearnableUpperProjection_52269751992477
// MI455X (gfx1250) — compile-verified, matrix operand order checked by reading, not run on hardware
//
#include <hip/hip_runtime.h>

// ---------------------------------------------------------------------------
// Types matching the gfx1250 WMMA builtin signatures
// ---------------------------------------------------------------------------
typedef __attribute__((ext_vector_type(16))) __bf16 v16bf;
typedef __attribute__((ext_vector_type(8)))  __bf16 v8bf;   // 16 bytes
typedef __attribute__((ext_vector_type(4)))  __bf16 v4bf;   // 8 bytes
typedef __attribute__((ext_vector_type(8)))  float  v8f;

#define LDS_STRIDE 40   // 32 K-elements + 8 bf16 pad (16B) to break bank conflicts

// ---------------------------------------------------------------------------
// gfx1250 async global->LDS copy (no VGPR staging, tracked by ASYNCcnt).
// VDST operand = 32-bit LDS byte address (low 32 bits of the generic shared
// pointer, per the flat-aperture rule).
// ---------------------------------------------------------------------------
__device__ __forceinline__ void async_load_b128(unsigned lds_addr,
                                                const void* gaddr) {
    asm volatile("global_load_async_to_lds_b128 %0, %1, off"
                 :: "v"(lds_addr), "v"((unsigned long long)(uintptr_t)gaddr)
                 : "memory");
}
__device__ __forceinline__ void wait_asynccnt0() {
    asm volatile("s_wait_asynccnt 0x0" ::: "memory");
}

// ---------------------------------------------------------------------------
// Load one 16x32 bf16 fragment (A or B operand) from an LDS tile.
// ISA layout (CDNA5 ISA, 16-bit A-matrix 16x32):
//   lanes 0-15 : row = lane,    K = 0-7  (VGPR0-3) and K = 16-23 (VGPR4-7)
//   lanes 16-31: row = lane-16, K = 8-15 (VGPR0-3) and K = 24-31 (VGPR4-7)
// => two contiguous 16-byte chunks per lane.
// ---------------------------------------------------------------------------
__device__ __forceinline__ v16bf load_frag_lds(const __bf16* tile, int row0) {
    const int lane = threadIdx.x & 31;
    const int r    = row0 + (lane & 15);
    const int kh   = (lane >> 4) << 3;              // 0 or 8
    const __bf16* p = tile + r * LDS_STRIDE;
    union { v16bf f; v8bf h[2]; } u;
    u.h[0] = *(const v8bf*)(p + kh);
    u.h[1] = *(const v8bf*)(p + kh + 16);
    return u.f;
}

// ---------------------------------------------------------------------------
// Tiled bf16 GEMM:  D[M,N] = A[M,K] * Bt[N,K]^T   (A, Bt row-major bf16)
// 256 threads = 8 waves; block tile 128(M) x 256(N); K-step 32;
// double-buffered LDS filled by ASYNC global->LDS B128 copies.
// Wave w: wm = (w&1)*64, wn = (w>>1)*64 -> 64x64 patch = 4x4 WMMA tiles.
// Per K-step: issue 6 async loads into the other buffer, load 8 fragments,
// run 16 v_wmma_f32_16x16x32_bf16 back-to-back, s_wait_asynccnt, barrier.
// ---------------------------------------------------------------------------
template <typename OutT>
__global__ __launch_bounds__(256)
void gemm_bf16_wmma(const __bf16* __restrict__ A,
                    const __bf16* __restrict__ Bt,
                    OutT* __restrict__ D,
                    int M, int N, int K)
{
    __shared__ __bf16 sA[2][128 * LDS_STRIDE];
    __shared__ __bf16 sB[2][256 * LDS_STRIDE];

    const int tid  = threadIdx.x;
    const int wave = tid >> 5;
    const int lane = tid & 31;
    const int bm   = blockIdx.x * 128;
    const int bn   = blockIdx.y * 256;
    const int wm   = (wave & 1) * 64;
    const int wn   = (wave >> 1) * 64;

    v8f acc[4][4];
    {
        v8f z = {};
        #pragma unroll
        for (int i = 0; i < 4; ++i)
            #pragma unroll
            for (int j = 0; j < 4; ++j) acc[i][j] = z;
    }

    // Cooperative staging. Tiles in 16B chunks: A = 128x32 bf16 = 512 chunks,
    // B = 256x32 bf16 = 1024 chunks. chunk c -> row c>>2, k-offset (c&3)*8.
    const int ar0 = tid >> 2,          ak0 = (tid & 3) * 8;
    const int ar1 = (tid + 256) >> 2,  ak1 = ((tid + 256) & 3) * 8;
    const int br0 = ar0, bk0 = ak0;
    const int br1 = ar1, bk1 = ak1;
    const int br2 = (tid + 512) >> 2,  bk2 = ((tid + 512) & 3) * 8;
    const int br3 = (tid + 768) >> 2,  bk3 = ((tid + 768) & 3) * 8;

    // Per-thread global source rows (byte step per K-step = 64B)
    const __bf16* gA0 = A  + (size_t)(bm + ar0) * K + ak0;
    const __bf16* gA1 = A  + (size_t)(bm + ar1) * K + ak1;
    const __bf16* gB0 = Bt + (size_t)(bn + br0) * K + bk0;
    const __bf16* gB1 = Bt + (size_t)(bn + br1) * K + bk1;
    const __bf16* gB2 = Bt + (size_t)(bn + br2) * K + bk2;
    const __bf16* gB3 = Bt + (size_t)(bn + br3) * K + bk3;

    // Per-thread LDS destinations, both buffers, as 32-bit LDS byte addresses.
    const unsigned dA0[2] = { (unsigned)(uintptr_t)&sA[0][ar0 * LDS_STRIDE + ak0],
                              (unsigned)(uintptr_t)&sA[1][ar0 * LDS_STRIDE + ak0] };
    const unsigned dA1[2] = { (unsigned)(uintptr_t)&sA[0][ar1 * LDS_STRIDE + ak1],
                              (unsigned)(uintptr_t)&sA[1][ar1 * LDS_STRIDE + ak1] };
    const unsigned dB0[2] = { (unsigned)(uintptr_t)&sB[0][br0 * LDS_STRIDE + bk0],
                              (unsigned)(uintptr_t)&sB[1][br0 * LDS_STRIDE + bk0] };
    const unsigned dB1[2] = { (unsigned)(uintptr_t)&sB[0][br1 * LDS_STRIDE + bk1],
                              (unsigned)(uintptr_t)&sB[1][br1 * LDS_STRIDE + bk1] };
    const unsigned dB2[2] = { (unsigned)(uintptr_t)&sB[0][br2 * LDS_STRIDE + bk2],
                              (unsigned)(uintptr_t)&sB[1][br2 * LDS_STRIDE + bk2] };
    const unsigned dB3[2] = { (unsigned)(uintptr_t)&sB[0][br3 * LDS_STRIDE + bk3],
                              (unsigned)(uintptr_t)&sB[1][br3 * LDS_STRIDE + bk3] };

    const int nk = K >> 5;     // K / 32

    // ---- preload K-step 0 into buffer 0 (async) -----------------------------
    async_load_b128(dA0[0], gA0);
    async_load_b128(dA1[0], gA1);
    async_load_b128(dB0[0], gB0);
    async_load_b128(dB1[0], gB1);
    async_load_b128(dB2[0], gB2);
    async_load_b128(dB3[0], gB3);
    wait_asynccnt0();

    for (int ks = 0; ks < nk; ++ks) {
        __syncthreads();               // all async fills + all reads of nb done
        const int cur = ks & 1;

        // ---- issue next K-step async copies into the other buffer -----------
        if (ks + 1 < nk) {
            const int nb = cur ^ 1;
            const int kb = (ks + 1) << 5;
            async_load_b128(dA0[nb], gA0 + kb);
            async_load_b128(dA1[nb], gA1 + kb);
            async_load_b128(dB0[nb], gB0 + kb);
            async_load_b128(dB1[nb], gB1 + kb);
            async_load_b128(dB2[nb], gB2 + kb);
            async_load_b128(dB3[nb], gB3 + kb);
        }
        if (ks + 2 < nk) {             // emits global_prefetch_b8 (L2 warm-up)
            const int kp = (ks + 2) << 5;
            __builtin_prefetch(gA0 + kp, 0, 1);
            __builtin_prefetch(gB0 + kp, 0, 1);
        }

        // ---- load ALL fragments, then 16 back-to-back WMMAs -----------------
        v16bf afr[4], bfr[4];
        #pragma unroll
        for (int i = 0; i < 4; ++i)
            afr[i] = load_frag_lds(sA[cur], wm + i * 16);
        #pragma unroll
        for (int i = 0; i < 4; ++i)
            bfr[i] = load_frag_lds(sB[cur], wn + i * 16);

        #pragma unroll
        for (int mt = 0; mt < 4; ++mt)
            #pragma unroll
            for (int nt = 0; nt < 4; ++nt)
                acc[mt][nt] = __builtin_amdgcn_wmma_f32_16x16x32_bf16(
                    false, afr[mt], false, bfr[nt], (short)0, acc[mt][nt],
                    false, false);

        // ---- make this wave's async fills complete before barrier -----------
        if (ks + 1 < nk) wait_asynccnt0();
    }

    // ---- writeback. C/D layout: VGPR r, lane l -> M = r + 8*(l>=16), N = l&15
    const int ln = lane & 15;
    const int lm = (lane >> 4) << 3;
    #pragma unroll
    for (int mt = 0; mt < 4; ++mt) {
        #pragma unroll
        for (int nt = 0; nt < 4; ++nt) {
            const int row0 = bm + wm + mt * 16 + lm;
            const int col  = bn + wn + nt * 16 + ln;
            #pragma unroll
            for (int r = 0; r < 8; ++r)
                D[(size_t)(row0 + r) * N + col] = (OutT)acc[mt][nt][r];
        }
    }
}

// ---------------------------------------------------------------------------
// Prep: f32 -> bf16 bulk convert, 4 elems/thread (float4 in, 8B out)
// ---------------------------------------------------------------------------
__global__ void cvt_f32_to_bf16_x4(const float* __restrict__ in,
                                   __bf16* __restrict__ out, int n4)
{
    int i = blockIdx.x * blockDim.x + threadIdx.x;
    if (i < n4) {
        float4 f = ((const float4*)in)[i];
        v4bf o = { (__bf16)f.x, (__bf16)f.y, (__bf16)f.z, (__bf16)f.w };
        ((v4bf*)out)[i] = o;
    }
}

// ---------------------------------------------------------------------------
// Prep: BsT[j][k] = B[k][j] * lambdas[j/2], bf16, via 32x32 LDS transpose
// ---------------------------------------------------------------------------
__global__ void build_bst_bf16(const float* __restrict__ B,
                               const float* __restrict__ lam,
                               __bf16* __restrict__ BsT, int K)
{
    __shared__ float tile[32][33];
    const int bx = blockIdx.x * 32;   // source column j
    const int by = blockIdx.y * 32;   // source row k
    const int tx = threadIdx.x, ty = threadIdx.y;   // (32, 8)
    #pragma unroll
    for (int i = 0; i < 32; i += 8)
        tile[ty + i][tx] = B[(size_t)(by + ty + i) * K + bx + tx];
    __syncthreads();
    #pragma unroll
    for (int i = 0; i < 32; i += 8) {
        const int j = bx + ty + i;
        const float s = lam[j >> 1];
        BsT[(size_t)j * K + by + tx] = (__bf16)(tile[tx][ty + i] * s);
    }
}

// ---------------------------------------------------------------------------
extern "C" void kernel_launch(void* const* d_in, const int* in_sizes, int n_in,
                              void* d_out, int out_size, void* d_ws, size_t ws_size,
                              hipStream_t stream)
{
    const float* x   = (const float*)d_in[0];   // (8192, 1024)
    const float* B   = (const float*)d_in[1];   // (1024, 1024)
    const float* C   = (const float*)d_in[2];   // (4096, 1024)
    const float* lam = (const float*)d_in[3];   // (512,)
    float* out = (float*)d_out;                 // (8192, 4096) f32

    const int BATCH = 8192, DIN = 1024, DOUT = 4096;

    char* ws = (char*)d_ws;
    __bf16* xbf  = (__bf16*)(ws);                                          // 16 MB
    __bf16* Cbf  = (__bf16*)(ws + (size_t)BATCH * DIN * 2);                //  8 MB
    __bf16* BsT  = (__bf16*)(ws + (size_t)(BATCH + DOUT) * DIN * 2);       //  2 MB
    __bf16* xbbf = (__bf16*)(ws + (size_t)(BATCH + DOUT + DIN) * DIN * 2); // 16 MB

    // prep
    cvt_f32_to_bf16_x4<<<(BATCH * DIN / 4 + 255) / 256, 256, 0, stream>>>(
        x, xbf, BATCH * DIN / 4);
    cvt_f32_to_bf16_x4<<<(DOUT * DIN / 4 + 255) / 256, 256, 0, stream>>>(
        C, Cbf, DOUT * DIN / 4);
    build_bst_bf16<<<dim3(DIN / 32, DIN / 32), dim3(32, 8), 0, stream>>>(
        B, lam, BsT, DIN);

    // GEMM1: xb = x @ (B*s)            (8192x1024) -> bf16 intermediate (L2-resident)
    gemm_bf16_wmma<__bf16><<<dim3(BATCH / 128, DIN / 256), 256, 0, stream>>>(
        xbf, BsT, xbbf, BATCH, DIN, DIN);

    // GEMM2: out = xb @ C^T            (8192x4096) -> f32 output
    gemm_bf16_wmma<float><<<dim3(BATCH / 128, DOUT / 256), 256, 0, stream>>>(
        xbbf, Cbf, out, BATCH, DOUT, DIN);
}
